// PicoDet_45483703664802
// MI455X (gfx1250) — matrix operand order repaired and checked by reading, not compiled, not run on hardware
//
#include <hip/hip_runtime.h>

// ---------------------------------------------------------------------------
// PicoDet head, fully fused per 16x16 tile, CDNA5 (gfx1250) wave32 WMMA.
// Async-LDS weight staging (GLOBAL_LOAD_ASYNC_TO_LDS_B128) when available.
// ---------------------------------------------------------------------------

typedef __attribute__((ext_vector_type(16))) _Float16     v16h;
typedef __attribute__((ext_vector_type(8)))  float        v8f;
typedef __attribute__((ext_vector_type(8)))  unsigned int v8u;
typedef __attribute__((ext_vector_type(4)))  int          v4i;

#define TPB 256

#if defined(__HIP_DEVICE_COMPILE__) && \
    __has_builtin(__builtin_amdgcn_global_load_async_to_lds_b128)
#define HAS_ASYNC_LDS 1
#else
#define HAS_ASYNC_LDS 0
#endif

static constexpr int WSTR = 104;   // LDS half-stride for 96-channel buffers (+8 pad)
static constexpr int YSTR = 56;    // LDS half-stride for 48-channel buffers (+8 pad)
static constexpr int R2 = 20, NP2 = 400;  // input region (halo 2)
static constexpr int R1 = 18, NP1 = 324;  // mid region   (halo 1)
static constexpr int NP0 = 256;           // output tile 16x16

// LDS layout (bytes)
static constexpr int off_Wg = 0;                          // ghost 1x1 weights, fp16 [4][48][WSTR]
static constexpr int off_Wh = off_Wg + 4 * 48 * WSTR * 2; // head weights fp16 [128][WSTR]
static constexpr int off_X  = off_Wh + 128 * WSTR * 2;    // input tile fp16 [400][WSTR]
static constexpr int off_Y  = off_X + NP2 * WSTR * 2;     // primary fp16 [400][YSTR] (aliased by rbuf)
static constexpr int off_C  = off_Y + NP2 * YSTR * 2;     // concat fp16 [324][WSTR]
static constexpr int off_s1 = off_C + NP1 * WSTR * 2;     // f32 [192]
static constexpr int off_b1 = off_s1 + 192 * 4;
static constexpr int off_s2 = off_b1 + 192 * 4;
static constexpr int off_b2 = off_s2 + 192 * 4;
static constexpr int off_dw = off_b2 + 192 * 4;           // dw weights f32 [4*48*9]
static constexpr int off_hb = off_dw + 1728 * 4;          // head bias f32 [128]
static constexpr int off_ob = off_hb + 128 * 4;           // obj logit f32 [256]
static constexpr int SMEM_BYTES = off_ob + 256 * 4;       // = 273472

// staging byte sizes/offsets for async weight copies (into dead X / C regions)
static constexpr int GW_BYTES  = 4 * 48 * 96 * 4;   // 73728 <= X region (83200)
static constexpr int CLS_BYTES = 80 * 96 * 4;       // 30720
static constexpr int OBJ_BYTES = 96 * 4;            // 384
static constexpr int REG_BYTES = 32 * 96 * 4;       // 12288 (total 43392 <= C region 67392)

__device__ __forceinline__ void wait_async0() {
#if HAS_ASYNC_LDS
#if __has_builtin(__builtin_amdgcn_s_wait_asynccnt)
  __builtin_amdgcn_s_wait_asynccnt(0);
#else
  asm volatile("s_wait_asynccnt 0" ::: "memory");
#endif
#endif
}

#if HAS_ASYNC_LDS
__device__ __forceinline__ void async_copy16(const void* gsrc, char* lds_dst) {
  __builtin_amdgcn_global_load_async_to_lds_b128(
      (__attribute__((address_space(1))) v4i*)gsrc,
      (__attribute__((address_space(3))) v4i*)lds_dst, 0, 0);
}
#endif

// --- A fragment: 16 rows (pixels) x 32 K, ISA 7.12.2 16-bit A layout --------
__device__ __forceinline__ v16h ld_fragA(const _Float16* p, int ldm, int row0,
                                         int kb, int nrows) {
  const int lane = threadIdx.x & 31;
  int r = row0 + (lane & 15);
  if (r > nrows - 1) r = nrows - 1;   // clamp padded rows (stores are guarded)
  const unsigned int* q =
      (const unsigned int*)(p + (size_t)r * ldm + kb + ((lane >> 4) << 3));
  union { v8u u; v16h h; } f;
#pragma unroll
  for (int v = 0; v < 4; ++v) {
    f.u[v]     = q[v];       // K = 8*half + 2v, 2v+1
    f.u[4 + v] = q[8 + v];   // K = 16 + 8*half + 2v, 2v+1
  }
  return f.h;
}

// --- B fragment: 32 K x 16 cols; weights stored [outCh][K] -------------------
__device__ __forceinline__ v16h ld_fragB(const _Float16* w, int ldw, int n0, int kb) {
  const int lane = threadIdx.x & 31;
  const int n = n0 + (lane & 15);
  const unsigned int* q =
      (const unsigned int*)(w + (size_t)n * ldw + kb + ((lane >> 4) << 3));
  union { v8u u; v16h h; } f;
#pragma unroll
  for (int v = 0; v < 4; ++v) {
    f.u[v]     = q[v];       // K = 8*half + 2v, 2v+1
    f.u[4 + v] = q[8 + v];   // K = 16 + 8*half + 2v, 2v+1
  }
  return f.h;
}

// --- generic tiled GEMM over 8 waves, K = 96 (3 x wmma 16x16x32) -------------
template <typename EPI>
__device__ __forceinline__ void gemm_tiles(const _Float16* A, int lda, int M,
                                           const _Float16* W, int ldw,
                                           int n0, int ntiles, EPI&& epi) {
  const int wave = threadIdx.x >> 5;
  const int mt = (M + 15) >> 4;
  for (int t = wave; t < mt * ntiles; t += TPB / 32) {
    const int mi = (t % mt) << 4;
    const int ni = n0 + ((t / mt) << 4);
    v8f acc = {0.f, 0.f, 0.f, 0.f, 0.f, 0.f, 0.f, 0.f};
#pragma unroll
    for (int kk = 0; kk < 96; kk += 32) {
      v16h a = ld_fragA(A, lda, mi, kk, M);
      v16h b = ld_fragB(W, ldw, ni, kk);
      acc = __builtin_amdgcn_wmma_f32_16x16x32_f16(
          false, a, false, b, (short)0, acc, false, false);
    }
    epi(mi, ni, acc);
  }
}

// ghost epilogue: +bias, relu6, zero out-of-map pixels (SAME-pad correctness).
// RI/MM are compile-time so /RI and %RI lower to constant mul+shift.
template <int RI, int MM>
__device__ __forceinline__ void ghost_store(v8f d, int mi, int ni,
                                            const float* bias, _Float16* O,
                                            int hb0, int wb0, int Hmap) {
  const int lane = threadIdx.x & 31;
  const int n = lane & 15;
  const float bb = bias[ni + n];
#pragma unroll
  for (int v = 0; v < 8; ++v) {
    const int m = mi + ((lane >> 4) << 3) + v;
    if (m < MM) {
      const int hh = hb0 + m / RI, ww = wb0 + m % RI;
      const bool inmap =
          ((unsigned)hh < (unsigned)Hmap) && ((unsigned)ww < (unsigned)Hmap);
      float x = fminf(fmaxf(d[v] + bb, 0.f), 6.f);
      O[(size_t)m * YSTR + ni + n] = inmap ? (_Float16)x : (_Float16)0.f;
    }
  }
}

// depthwise 3x3 + BN + relu6 into concat cols [48..95]; copy primary into [0..47]
template <int RI, int RO>
__device__ __forceinline__ void dw_concat(const _Float16* Y, _Float16* Cc,
                                          const float* w9, const float* b2) {
  constexpr int total = RO * RO * 48;
  for (int i = threadIdx.x; i < total; i += TPB) {
    const int c = i % 48;
    const int p = i / 48;
    const int oy = p / RO, ox = p % RO;
    const _Float16* yp = Y + ((size_t)oy * RI + ox) * YSTR + c;  // top-left tap
    float acc = b2[c];
#pragma unroll
    for (int ky = 0; ky < 3; ++ky)
#pragma unroll
      for (int kx = 0; kx < 3; ++kx)
        acc += w9[c * 9 + ky * 3 + kx] * (float)yp[(ky * RI + kx) * YSTR];
    acc = fminf(fmaxf(acc, 0.f), 6.f);
    Cc[(size_t)p * WSTR + 48 + c] = (_Float16)acc;
    Cc[(size_t)p * WSTR + c] = yp[(RI + 1) * YSTR];  // center primary value
  }
}

template <int LVL>
__global__ __launch_bounds__(TPB) void picodet_level_kernel(
    const float* __restrict__ fin, const float* __restrict__ gw1,
    const float* __restrict__ g1g, const float* __restrict__ g1b,
    const float* __restrict__ g1m, const float* __restrict__ g1v,
    const float* __restrict__ gdw, const float* __restrict__ g2g,
    const float* __restrict__ g2b, const float* __restrict__ g2m,
    const float* __restrict__ g2v, const float* __restrict__ cwp,
    const float* __restrict__ cbp, const float* __restrict__ owp,
    const float* __restrict__ obp, const float* __restrict__ rwp,
    const float* __restrict__ rbp, float* __restrict__ out) {
  constexpr int H = (LVL == 0) ? 64 : (LVL == 1) ? 32 : 16;
  constexpr float STRIDE = (float)(8 << LVL);
  constexpr int ABASE = (LVL == 0) ? 0 : (LVL == 1) ? 4096 : 5120;
  constexpr int TILES = H / 16;

  extern __shared__ char smem[];
  _Float16* Wg   = (_Float16*)(smem + off_Wg);
  _Float16* Wh   = (_Float16*)(smem + off_Wh);
  _Float16* X    = (_Float16*)(smem + off_X);
  _Float16* Ybuf = (_Float16*)(smem + off_Y);
  _Float16* Cc   = (_Float16*)(smem + off_C);
  float* s1  = (float*)(smem + off_s1);
  float* b1s = (float*)(smem + off_b1);
  float* s2  = (float*)(smem + off_s2);
  float* b2s = (float*)(smem + off_b2);
  float* dwW = (float*)(smem + off_dw);
  float* hb  = (float*)(smem + off_hb);
  float* obuf = (float*)(smem + off_ob);
  float* rbuf = (float*)(smem + off_Y);  // alias: Ybuf dead when reg head runs

  const int tid = threadIdx.x;
  const int tpi = TILES * TILES;
  const int b  = blockIdx.x / tpi;
  const int t  = blockIdx.x % tpi;
  const int h0 = (t / TILES) * 16;
  const int w0 = (t % TILES) * 16;

#if HAS_ASYNC_LDS
  // ---- stage raw f32 weights into (currently dead) X / C regions ----------
  {
    const char* gsrc = (const char*)gw1;
    for (int off = tid * 16; off < GW_BYTES; off += TPB * 16)
      async_copy16(gsrc + off, smem + off_X + off);
    const char* csrc = (const char*)cwp + (size_t)LVL * CLS_BYTES;
    for (int off = tid * 16; off < CLS_BYTES; off += TPB * 16)
      async_copy16(csrc + off, smem + off_C + off);
    if (tid * 16 < OBJ_BYTES)
      async_copy16((const char*)owp + (size_t)LVL * OBJ_BYTES + tid * 16,
                   smem + off_C + CLS_BYTES + tid * 16);
    const char* rsrc = (const char*)rwp + (size_t)LVL * REG_BYTES;
    for (int off = tid * 16; off < REG_BYTES; off += TPB * 16)
      async_copy16(rsrc + off, smem + off_C + CLS_BYTES + OBJ_BYTES + off);
  }
#else
  if (tid < 8) {  // gfx1250 global_prefetch_b8 on shared weights
    __builtin_prefetch(gw1 + tid * 2048, 0, 0);
    __builtin_prefetch(cwp + tid * 960, 0, 0);
  }
#endif

  // ---- fold BN into scales/biases (overlaps with async staging) -----------
  for (int i = tid; i < 192; i += TPB) {
    float sa = g1g[i] * rsqrtf(g1v[i] + 1e-5f);
    s1[i] = sa;
    b1s[i] = g1b[i] - g1m[i] * sa;
    float sb = g2g[i] * rsqrtf(g2v[i] + 1e-5f);
    s2[i] = sb;
    b2s[i] = g2b[i] - g2m[i] * sb;
  }
  wait_async0();
  __syncthreads();

  // ---- convert weights (BN-folded fp16) into Wg/Wh ------------------------
#if HAS_ASYNC_LDS
  const float* stgW = (const float*)(smem + off_X);
  const float* stgH = (const float*)(smem + off_C);
#endif
  for (int i = tid; i < 4 * 48 * 96; i += TPB) {
    int go = i / 96, k = i % 96;
#if HAS_ASYNC_LDS
    float wv = stgW[i];
#else
    float wv = gw1[i];
#endif
    Wg[(size_t)go * WSTR + k] = (_Float16)(wv * s1[go]);
  }
  for (int i = tid; i < 128 * 96; i += TPB) {
    int o = i / 96, k = i % 96;
    float wv = 0.f;
#if HAS_ASYNC_LDS
    if (o < 80)       wv = stgH[o * 96 + k];
    else if (o == 80) wv = stgH[80 * 96 + k];
    else if (o >= 96) wv = stgH[81 * 96 + (o - 96) * 96 + k];
#else
    if (o < 80)       wv = cwp[((size_t)LVL * 80 + o) * 96 + k];
    else if (o == 80) wv = owp[(size_t)LVL * 96 + k];
    else if (o >= 96) wv = rwp[((size_t)LVL * 32 + (o - 96)) * 96 + k];
#endif
    Wh[(size_t)o * WSTR + k] = (_Float16)wv;
  }
  for (int i = tid; i < 4 * 48 * 9; i += TPB) dwW[i] = gdw[i] * s2[i / 9];
  for (int i = tid; i < 128; i += TPB) {
    float bv = 0.f;
    if (i < 80)       bv = cbp[LVL * 80 + i];
    else if (i == 80) bv = obp[LVL];
    else if (i >= 96) bv = rbp[LVL * 32 + i - 96];
    hb[i] = bv;
  }
  __syncthreads();  // converts done before X tile overwrites staging

  // ---- stage input tile (fp16, zero-padded halo) --------------------------
  for (int i = tid; i < NP2 * 96; i += TPB) {
    int c = i / NP2, p = i % NP2;
    int hh = h0 - 2 + p / R2, ww = w0 - 2 + p % R2;
    float xv = 0.f;
    if ((unsigned)hh < (unsigned)H && (unsigned)ww < (unsigned)H)
      xv = fin[(((size_t)b * 96 + c) * H + hh) * H + ww];
    X[(size_t)p * WSTR + c] = (_Float16)xv;
  }
  __syncthreads();

  // ---- two towers: ghosts {0,1} -> cls/obj head, ghosts {2,3} -> reg head --
  for (int tower = 0; tower < 2; ++tower) {
    const int g0 = tower * 2;

    // ghost-A primary 1x1: X[400x96] x W^T -> Y[400x48]
    {
      const float* bias = b1s + g0 * 48;
      gemm_tiles(X, WSTR, NP2, Wg + (size_t)g0 * 48 * WSTR, WSTR, 0, 3,
                 [&](int mi, int ni, v8f d) {
                   ghost_store<R2, NP2>(d, mi, ni, bias, Ybuf, h0 - 2, w0 - 2, H);
                 });
    }
    __syncthreads();
    dw_concat<R2, R1>(Ybuf, Cc, dwW + g0 * 48 * 9, b2s + g0 * 48);
    __syncthreads();

    // ghost-B primary 1x1: C[324x96] -> Y[324x48]
    {
      const float* bias = b1s + (g0 + 1) * 48;
      gemm_tiles(Cc, WSTR, NP1, Wg + (size_t)(g0 + 1) * 48 * WSTR, WSTR, 0, 3,
                 [&](int mi, int ni, v8f d) {
                   ghost_store<R1, NP1>(d, mi, ni, bias, Ybuf, h0 - 1, w0 - 1, H);
                 });
    }
    __syncthreads();
    dw_concat<R1, 16>(Ybuf, Cc, dwW + (g0 + 1) * 48 * 9, b2s + (g0 + 1) * 48);
    __syncthreads();

    if (tower == 0) {
      // objectness: rows [80..95] of Wh (only channel 80 real) -> obuf
      gemm_tiles(Cc, WSTR, NP0, Wh, WSTR, 80, 1, [&](int mi, int ni, v8f d) {
        const int lane = threadIdx.x & 31;
        if ((lane & 15) == 0) {
#pragma unroll
          for (int v = 0; v < 8; ++v)
            obuf[mi + ((lane >> 4) << 3) + v] = d[v] + hb[80];
        }
      });
      __syncthreads();
      // class scores: sigmoid(cls + obj) straight to global output
      gemm_tiles(Cc, WSTR, NP0, Wh, WSTR, 0, 5, [&](int mi, int ni, v8f d) {
        const int lane = threadIdx.x & 31;
        const int n = lane & 15;
        const float bb = hb[ni + n];
#pragma unroll
        for (int v = 0; v < 8; ++v) {
          const int m = mi + ((lane >> 4) << 3) + v;
          const float x = d[v] + bb + obuf[m];
          const float sc = 1.f / (1.f + __expf(-x));
          const int hh = h0 + (m >> 4), ww = w0 + (m & 15);
          const size_t row = (size_t)b * 5376 + ABASE + hh * H + ww;
          out[row * 84 + 4 + ni + n] = sc;
        }
      });
      __syncthreads();
    } else {
      // regression logits -> rbuf (f32, aliases Ybuf)
      gemm_tiles(Cc, WSTR, NP0, Wh, WSTR, 96, 2, [&](int mi, int ni, v8f d) {
        const int lane = threadIdx.x & 31;
        const int n = lane & 15;
        const float bb = hb[ni + n];
#pragma unroll
        for (int v = 0; v < 8; ++v) {
          const int m = mi + ((lane >> 4) << 3) + v;
          rbuf[m * 32 + (ni - 96) + n] = d[v] + bb;
        }
      });
      __syncthreads();
      // DFL expectation + box decode, one pixel per thread
      if (tid < NP0) {
        const float* r = rbuf + tid * 32;
        const int hh = h0 + (tid >> 4), ww = w0 + (tid & 15);
        float ltrb[4];
#pragma unroll
        for (int g = 0; g < 4; ++g) {
          float mx = r[g * 8];
#pragma unroll
          for (int j = 1; j < 8; ++j) mx = fmaxf(mx, r[g * 8 + j]);
          float s = 0.f, e = 0.f;
#pragma unroll
          for (int j = 0; j < 8; ++j) {
            float t2 = __expf(r[g * 8 + j] - mx);
            s += t2;
            e += t2 * (float)j;
          }
          ltrb[g] = e / s * STRIDE;
        }
        const float ax = (ww + 0.5f) * STRIDE;
        const float ay = (hh + 0.5f) * STRIDE;
        const size_t row = ((size_t)b * 5376 + ABASE + hh * H + ww) * 84;
        out[row + 0] = ax - ltrb[0];
        out[row + 1] = ay - ltrb[1];
        out[row + 2] = ax + ltrb[2];
        out[row + 3] = ay + ltrb[3];
      }
      __syncthreads();
    }
  }
}

extern "C" void kernel_launch(void* const* d_in, const int* in_sizes, int n_in,
                              void* d_out, int out_size, void* d_ws,
                              size_t ws_size, hipStream_t stream) {
  const float* f3  = (const float*)d_in[0];
  const float* f4  = (const float*)d_in[1];
  const float* f5  = (const float*)d_in[2];
  const float* gw1 = (const float*)d_in[3];
  const float* g1g = (const float*)d_in[4];
  const float* g1b = (const float*)d_in[5];
  const float* g1m = (const float*)d_in[6];
  const float* g1v = (const float*)d_in[7];
  const float* gdw = (const float*)d_in[8];
  const float* g2g = (const float*)d_in[9];
  const float* g2b = (const float*)d_in[10];
  const float* g2m = (const float*)d_in[11];
  const float* g2v = (const float*)d_in[12];
  const float* cw  = (const float*)d_in[13];
  const float* cb  = (const float*)d_in[14];
  const float* ow  = (const float*)d_in[15];
  const float* ob  = (const float*)d_in[16];
  const float* rw  = (const float*)d_in[17];
  const float* rb  = (const float*)d_in[18];
  float* out = (float*)d_out;

  (void)hipFuncSetAttribute((const void*)&picodet_level_kernel<0>,
                            hipFuncAttributeMaxDynamicSharedMemorySize,
                            SMEM_BYTES);
  (void)hipFuncSetAttribute((const void*)&picodet_level_kernel<1>,
                            hipFuncAttributeMaxDynamicSharedMemorySize,
                            SMEM_BYTES);
  (void)hipFuncSetAttribute((const void*)&picodet_level_kernel<2>,
                            hipFuncAttributeMaxDynamicSharedMemorySize,
                            SMEM_BYTES);

  picodet_level_kernel<0><<<32 * 16, TPB, SMEM_BYTES, stream>>>(
      f3, gw1, g1g, g1b, g1m, g1v, gdw, g2g, g2b, g2m, g2v, cw, cb, ow, ob, rw,
      rb, out);
  picodet_level_kernel<1><<<32 * 4, TPB, SMEM_BYTES, stream>>>(
      f4, gw1, g1g, g1b, g1m, g1v, gdw, g2g, g2b, g2m, g2v, cw, cb, ow, ob, rw,
      rb, out);
  picodet_level_kernel<2><<<32 * 1, TPB, SMEM_BYTES, stream>>>(
      f5, gw1, g1g, g1b, g1m, g1v, gdw, g2g, g2b, g2m, g2v, cw, cb, ow, ob, rw,
      rb, out);
}
